// DampingGCN_53515292508330
// MI455X (gfx1250) — matrix operand order repaired and checked by reading, not compiled, not run on hardware
//
#include <hip/hip_runtime.h>
#include <stdint.h>
#include <stddef.h>

// ---------------------------------------------------------------------------
// CDNA5 (gfx1250) DampingGCN pipeline.
// Dense GEMMs -> v_wmma_f32_16x16x32_bf16 (bf16 in, f32 accumulate).
//   - feature tensors kept in bf16 end-to-end for GEMM inputs (halves HBM
//     traffic on the A side); A staging is a pure 16B copy into 4KB LDS
//   - B read directly from pre-transposed bf16 weights (32B contiguous/lane,
//     L2-resident) -> no B LDS staging
// Aggregation path (GEMM out -> atomics) stays f32 for accumulation precision.
// Sparse GCN aggregation -> global_atomic_add_f32 (bandwidth bound).
// ---------------------------------------------------------------------------

typedef __attribute__((ext_vector_type(16))) __bf16          v16bf;
typedef __attribute__((ext_vector_type(16))) unsigned short  v16u;
typedef __attribute__((ext_vector_type(8)))  float           v8f;

__device__ __forceinline__ uint16_t f2bf(float f) {
  union { float f; uint32_t u; } v; v.f = f;
  uint32_t r = v.u + 0x7FFFu + ((v.u >> 16) & 1u);   // round-to-nearest-even
  return (uint16_t)(r >> 16);
}
__device__ __forceinline__ uint32_t pack2bf(float a, float b) {
  return (uint32_t)f2bf(a) | ((uint32_t)f2bf(b) << 16);
}
__device__ __forceinline__ float sigmoidf_(float v) { return 1.0f / (1.0f + __expf(-v)); }

// --------------------------- small utility kernels -------------------------

__global__ void k_zero_f32(float* __restrict__ p, int n) {
  int i = blockIdx.x * 256 + threadIdx.x;
  if (i < n) p[i] = 0.0f;
}

__global__ void k_deg_accum(const int* __restrict__ dst, float* __restrict__ deg, int E) {
  int i = blockIdx.x * 256 + threadIdx.x;
  if (i < E) atomicAdd(&deg[dst[i]], 1.0f);
}

__global__ void k_deg_to_dinv(float* __restrict__ d, int n) {
  int i = blockIdx.x * 256 + threadIdx.x;
  if (i < n) d[i] = rsqrtf(1.0f + d[i]);     // deg includes self-loop
}

// W[k][n] (f32, row-major [K,NOUT]) -> Wt[n][k] (bf16, [NOUT,K]) transposed
__global__ void k_conv_bf16_t(const float* __restrict__ W, uint16_t* __restrict__ o,
                              int K, int NOUT) {
  int i = blockIdx.x * 256 + threadIdx.x;
  if (i < K * NOUT) {
    int k = i / NOUT, n = i - k * NOUT;      // coalesced read
    o[(size_t)n * K + k] = f2bf(W[i]);       // scattered one-time write
  }
}

// F[2i],F[2i+1] = bf16(relu(G[2i])), bf16(relu(G[2i+1]))  -- packed u32 store
__global__ void k_relu_bf16(const float* __restrict__ G, uint16_t* __restrict__ F,
                            long long n2) {
  long long i = (long long)blockIdx.x * 256 + threadIdx.x;
  if (i < n2) {
    float2 v = ((const float2*)G)[i];
    ((uint32_t*)F)[i] = pack2bf(fmaxf(v.x, 0.f), fmaxf(v.y, 0.f));
  }
}

// state = bf16(sigmoid(x @ se_W + se_b)), x: [N,3], W: [3,128]
__global__ void k_encode_state(const float* __restrict__ x, const float* __restrict__ W,
                               const float* __restrict__ b, uint16_t* __restrict__ out, int N) {
  long long idx = (long long)blockIdx.x * 256 + threadIdx.x;
  if (idx >= (long long)N * 128) return;
  int i = (int)(idx >> 7), f = (int)(idx & 127);
  float v = x[i * 3 + 0] * W[f] + x[i * 3 + 1] * W[128 + f] + x[i * 3 + 2] * W[256 + f] + b[f];
  out[idx] = f2bf(sigmoidf_(v));
}

// phys = bf16(sigmoid([alpha,torque] @ pe_W1 + pe_b1)), W: [2,128]
__global__ void k_encode_phys(const float* __restrict__ a, const float* __restrict__ t,
                              const float* __restrict__ W, const float* __restrict__ b,
                              uint16_t* __restrict__ out, int N) {
  long long idx = (long long)blockIdx.x * 256 + threadIdx.x;
  if (idx >= (long long)N * 128) return;
  int i = (int)(idx >> 7), f = (int)(idx & 127);
  out[idx] = f2bf(sigmoidf_(a[i] * W[f] + t[i] * W[128 + f] + b[f]));
}

// G[i,f] = dinv[i]^2 * T[i,f] + bias[f]   (self-loop contribution, initializes G)
__global__ void k_gcn_self_bias(const float* __restrict__ T, const float* __restrict__ dinv,
                                const float* __restrict__ bias, float* __restrict__ G, int N) {
  long long idx = (long long)blockIdx.x * 256 + threadIdx.x;
  if (idx >= (long long)N * 128) return;
  int i = (int)(idx >> 7), f = (int)(idx & 127);
  float di = dinv[i];
  G[idx] = di * di * T[idx] + bias[f];
}

// one wave per edge: AGG[dst] += dinv[src]*dinv[dst] * T[src]   (128 floats via lane float4)
__global__ void k_gcn_edge_agg(const int* __restrict__ src, const int* __restrict__ dst,
                               const float* __restrict__ dinv, const float* __restrict__ T,
                               float* __restrict__ AGG, int E) {
  int e = blockIdx.x * 8 + (threadIdx.x >> 5);
  if (e >= E) return;
  int lane = threadIdx.x & 31;
  int s = src[e], d = dst[e];
  float nrm = dinv[s] * dinv[d];
  const float4* hs = (const float4*)(T + (size_t)s * 128);
  float4 v = hs[lane];
  float* out = AGG + (size_t)d * 128 + lane * 4;
  atomicAdd(out + 0, nrm * v.x);
  atomicAdd(out + 1, nrm * v.y);
  atomicAdd(out + 2, nrm * v.z);
  atomicAdd(out + 3, nrm * v.w);
}

// out[n] = sigmoid(D2[n,:] @ W3 + b3), D2: [N,64] f32, wave per node
__global__ void k_dp3(const float* __restrict__ D2, const float* __restrict__ W3,
                      const float* __restrict__ b3, float* __restrict__ out, int N) {
  int n = blockIdx.x * 8 + (threadIdx.x >> 5);
  if (n >= N) return;
  int lane = threadIdx.x & 31;
  float s = D2[(size_t)n * 64 + lane] * W3[lane] +
            D2[(size_t)n * 64 + 32 + lane] * W3[32 + lane];
  #pragma unroll
  for (int off = 16; off > 0; off >>= 1) s += __shfl_xor(s, off, 32);
  if (lane == 0) out[n] = sigmoidf_(s + b3[0]);
}

// ------------------------------ WMMA GEMM ----------------------------------
// Y[Nrows, NOUT] = concat(X1[:,:K1], X2[:,:K2]) @ W[K1+K2, NOUT]  (+bias, act)
// X1/X2 are bf16 [Nrows, K*]; Wt is bf16 TRANSPOSED weight [NOUT, K].
// NOUT/ACT/OUTBF compile-time -> zero control flow around v_wmma.
// Block: 256 thr (8 waves): 4 M-tiles x 2 N-groups; NJ = NOUT/32 tiles/wave.
// K1, K2 must be multiples of 32 (K2 may be 0).
template <int NOUT, int ACT, int OUTBF>
__global__ void __launch_bounds__(256)
k_gemm_bf16_wmma(const uint16_t* __restrict__ X1, int K1,
                 const uint16_t* __restrict__ X2, int K2,
                 const uint16_t* __restrict__ Wt,
                 const float* __restrict__ bias,
                 void* __restrict__ Yv, int Nrows) {
  constexpr int NJ = NOUT / 32;            // N-tiles per wave
  __shared__ uint16_t sA[64 * 32];         // A chunk, row-major [row][k], bf16

  float*    Yf = (float*)Yv;
  uint16_t* Yh = (uint16_t*)Yv;

  const int tid  = threadIdx.x;
  const int lane = tid & 31;
  const int wid  = __builtin_amdgcn_readfirstlane(tid >> 5);  // wave-uniform
  const int rowBase = blockIdx.x * 64;
  const int K = K1 + K2;

  const int m0     = (wid & 3) * 16;              // wave's 16-row strip
  const int n0base = (wid >> 2) * (NOUT / 2);     // wave's N-group origin

  v8f acc[NJ];
  #pragma unroll
  for (int j = 0; j < NJ; ++j)
    #pragma unroll
    for (int r = 0; r < 8; ++r) acc[j][r] = 0.0f;

  // per-lane fragment geometry (constant across K chunks)
  const int arow = m0 + (lane & 15);
  const int half = lane >> 4;                     // A: K-half select
  const int kb   = half << 4;                     // B: lanes 0-15 K0..15, 16-31 K16..31

  // A staging geometry: 64 rows x 32 cols bf16 = 4KB = 256 thr x 16B
  const int cr  = tid >> 2;                       // staging row
  const int cc8 = (tid & 3) << 3;                 // staging col (8 bf16 = 16B)

  for (int k0 = 0; k0 < K; k0 += 32) {
    // ---- scalar (uniform) split-K source select: whole chunk is in X1 or X2
    const uint16_t* Xp;
    int Kst, ko;
    if (k0 < K1) { Xp = X1; Kst = K1; ko = k0; }
    else         { Xp = X2; Kst = K2; ko = k0 - K1; }

    // ---- stage A: pure 16B copy (global_load_b128 -> ds_store_b128)
    {
      int gr = rowBase + cr;
      uint4 v = make_uint4(0u, 0u, 0u, 0u);
      if (gr < Nrows) v = *(const uint4*)(Xp + (size_t)gr * Kst + ko + cc8);
      *(uint4*)&sA[cr * 32 + cc8] = v;
    }
    // prefetch next A chunk into GL2 (global_prefetch_b8)
    if (k0 + 32 < K1) __builtin_prefetch(&X1[(size_t)(rowBase + cr) * K1 + k0 + 32], 0, 1);
    __syncthreads();

    // ---- A fragment, K order matching B (contiguous):
    //  lanes 0-15: row=lane,    elems {0..15}->K0..15
    //  lanes16-31: row=lane-16, elems {0..15}->K16..31
    v16u au;
    #pragma unroll
    for (int e = 0; e < 16; ++e) {
      int k = half * 16 + e;
      au[e] = sA[arow * 32 + k];
    }
    v16bf a = __builtin_bit_cast(v16bf, au);

    // ---- B fragments straight from transposed global weights (32B/lane,
    //      L2-resident) + WMMA chain; no branches: EXEC all-ones
    #pragma unroll
    for (int j = 0; j < NJ; ++j) {
      const int bn = n0base + j * 16 + (lane & 15);
      const uint16_t* bp = Wt + (size_t)bn * K + k0 + kb;
      v16u bu;
      #pragma unroll
      for (int e = 0; e < 16; ++e) bu[e] = bp[e];
      v16bf b = __builtin_bit_cast(v16bf, bu);
      acc[j] = __builtin_amdgcn_wmma_f32_16x16x32_bf16(
          false, a, false, b, (short)0, acc[j], false, false);
    }
    __syncthreads();
  }

  // ---- epilogue: C layout VGPR r -> M = r + 8*(lane>=16), N = lane%16
  #pragma unroll
  for (int j = 0; j < NJ; ++j) {
    int col = n0base + j * 16 + (lane & 15);
    float bv = bias ? bias[col] : 0.0f;
    #pragma unroll
    for (int r = 0; r < 8; ++r) {
      int gr = rowBase + m0 + r + (half << 3);
      if (gr < Nrows) {
        float v = acc[j][r] + bv;
        if (ACT == 1) v = sigmoidf_(v);
        if (OUTBF) Yh[(size_t)gr * NOUT + col] = f2bf(v);
        else       Yf[(size_t)gr * NOUT + col] = v;
      }
    }
  }
}

// ------------------------------- launcher ----------------------------------

extern "C" void kernel_launch(void* const* d_in, const int* in_sizes, int n_in,
                              void* d_out, int out_size, void* d_ws, size_t ws_size,
                              hipStream_t stream) {
  const float* x      = (const float*)d_in[0];
  const float* alpha  = (const float*)d_in[1];
  const float* torque = (const float*)d_in[2];
  const int*   ei     = (const int*)  d_in[3];
  const float* se_W  = (const float*)d_in[4];
  const float* se_b  = (const float*)d_in[5];
  const float* pe_W1 = (const float*)d_in[6];
  const float* pe_b1 = (const float*)d_in[7];
  const float* pe_W2 = (const float*)d_in[8];
  const float* pe_b2 = (const float*)d_in[9];
  const float* c1_W  = (const float*)d_in[10];
  const float* c1_b  = (const float*)d_in[11];
  const float* c2_W  = (const float*)d_in[12];
  const float* c2_b  = (const float*)d_in[13];
  const float* p1_W  = (const float*)d_in[14];
  const float* p1_b  = (const float*)d_in[15];
  const float* p2_W  = (const float*)d_in[16];
  const float* p2_b  = (const float*)d_in[17];
  const float* cc_W  = (const float*)d_in[18];
  const float* cc_b  = (const float*)d_in[19];
  const float* dp_W1 = (const float*)d_in[20];
  const float* dp_b1 = (const float*)d_in[21];
  const float* dp_W2 = (const float*)d_in[22];
  const float* dp_b2 = (const float*)d_in[23];
  const float* dp_W3 = (const float*)d_in[24];
  const float* dp_b3 = (const float*)d_in[25];

  const int N = in_sizes[0] / 3;
  const int E = in_sizes[3] / 2;
  const int* src = ei;
  const int* dst = ei + E;

  // ---- workspace layout (256B-aligned carve-out)
  char* ws = (char*)d_ws;
  auto carve = [&](size_t bytes) -> char* {
    char* p = ws;
    ws += (bytes + 255) & ~(size_t)255;
    return p;
  };
  float*    dinv = (float*)carve((size_t)N * 4);
  float*    T  = (float*)carve((size_t)N * 128 * 4);     // GEMM out (f32)
  float*    G  = (float*)carve((size_t)N * 128 * 4);     // aggregation (f32)
  uint16_t* F0 = (uint16_t*)carve((size_t)N * 128 * 2);  // bf16 features
  uint16_t* F1 = (uint16_t*)carve((size_t)N * 128 * 2);
  uint16_t* F2 = (uint16_t*)carve((size_t)N * 128 * 2);
  uint16_t* w_pe2 = (uint16_t*)carve(128 * 128 * 2);     // transposed [NOUT,K]
  uint16_t* w_c1  = (uint16_t*)carve(128 * 128 * 2);
  uint16_t* w_c2  = (uint16_t*)carve(128 * 128 * 2);
  uint16_t* w_p1  = (uint16_t*)carve(128 * 128 * 2);
  uint16_t* w_p2  = (uint16_t*)carve(128 * 128 * 2);
  uint16_t* w_cc  = (uint16_t*)carve(256 * 128 * 2);
  uint16_t* w_dp1 = (uint16_t*)carve(128 * 128 * 2);
  uint16_t* w_dp2 = (uint16_t*)carve(128 * 64 * 2);

  const long long NF = (long long)N * 128;
  const int gNF   = (int)((NF + 255) / 256);
  const int gNF2  = (int)((NF / 2 + 255) / 256);
  const int gN    = (N + 255) / 256;
  const int gE    = (E + 255) / 256;
  const int gEdge = (E + 7) / 8;
  const int gGemm = (N + 63) / 64;
  dim3 blk(256);

  // ---- degree / normalization
  k_zero_f32<<<gN, blk, 0, stream>>>(dinv, N);
  k_deg_accum<<<gE, blk, 0, stream>>>(dst, dinv, E);
  k_deg_to_dinv<<<gN, blk, 0, stream>>>(dinv, N);

  // ---- weights -> bf16, transposed to [NOUT, K]
  k_conv_bf16_t<<<64,  blk, 0, stream>>>(pe_W2, w_pe2, 128, 128);
  k_conv_bf16_t<<<64,  blk, 0, stream>>>(c1_W,  w_c1,  128, 128);
  k_conv_bf16_t<<<64,  blk, 0, stream>>>(c2_W,  w_c2,  128, 128);
  k_conv_bf16_t<<<64,  blk, 0, stream>>>(p1_W,  w_p1,  128, 128);
  k_conv_bf16_t<<<64,  blk, 0, stream>>>(p2_W,  w_p2,  128, 128);
  k_conv_bf16_t<<<128, blk, 0, stream>>>(cc_W,  w_cc,  256, 128);
  k_conv_bf16_t<<<64,  blk, 0, stream>>>(dp_W1, w_dp1, 128, 128);
  k_conv_bf16_t<<<32,  blk, 0, stream>>>(dp_W2, w_dp2, 128, 64);

  // ---- state path: encode -> GCN(c1) -> GCN(c2); state final in F0
  k_encode_state<<<gNF, blk, 0, stream>>>(x, se_W, se_b, F0, N);
  k_gemm_bf16_wmma<128, 0, 0><<<gGemm, blk, 0, stream>>>(F0, 128, nullptr, 0, w_c1, nullptr, T, N);
  k_gcn_self_bias<<<gNF, blk, 0, stream>>>(T, dinv, c1_b, G, N);
  k_gcn_edge_agg<<<gEdge, blk, 0, stream>>>(src, dst, dinv, T, G, E);
  k_relu_bf16<<<gNF2, blk, 0, stream>>>(G, F1, NF / 2);
  k_gemm_bf16_wmma<128, 0, 0><<<gGemm, blk, 0, stream>>>(F1, 128, nullptr, 0, w_c2, nullptr, T, N);
  k_gcn_self_bias<<<gNF, blk, 0, stream>>>(T, dinv, c2_b, G, N);
  k_gcn_edge_agg<<<gEdge, blk, 0, stream>>>(src, dst, dinv, T, G, E);
  k_relu_bf16<<<gNF2, blk, 0, stream>>>(G, F0, NF / 2);

  // ---- phys path: encode -> pe2(sigmoid,bf16) -> GCN(p1) -> GCN(p2); final F2
  k_encode_phys<<<gNF, blk, 0, stream>>>(alpha, torque, pe_W1, pe_b1, F1, N);
  k_gemm_bf16_wmma<128, 1, 1><<<gGemm, blk, 0, stream>>>(F1, 128, nullptr, 0, w_pe2, pe_b2, F2, N);
  k_gemm_bf16_wmma<128, 0, 0><<<gGemm, blk, 0, stream>>>(F2, 128, nullptr, 0, w_p1, nullptr, T, N);
  k_gcn_self_bias<<<gNF, blk, 0, stream>>>(T, dinv, p1_b, G, N);
  k_gcn_edge_agg<<<gEdge, blk, 0, stream>>>(src, dst, dinv, T, G, E);
  k_relu_bf16<<<gNF2, blk, 0, stream>>>(G, F1, NF / 2);
  k_gemm_bf16_wmma<128, 0, 0><<<gGemm, blk, 0, stream>>>(F1, 128, nullptr, 0, w_p2, nullptr, T, N);
  k_gcn_self_bias<<<gNF, blk, 0, stream>>>(T, dinv, p2_b, G, N);
  k_gcn_edge_agg<<<gEdge, blk, 0, stream>>>(src, dst, dinv, T, G, E);
  k_relu_bf16<<<gNF2, blk, 0, stream>>>(G, F2, NF / 2);

  // ---- combined GCN: concat(state=F0, phys=F2) @ cc_W via split-K GEMM
  k_gemm_bf16_wmma<128, 0, 0><<<gGemm, blk, 0, stream>>>(F0, 128, F2, 128, w_cc, nullptr, T, N);
  k_gcn_self_bias<<<gNF, blk, 0, stream>>>(T, dinv, cc_b, G, N);
  k_gcn_edge_agg<<<gEdge, blk, 0, stream>>>(src, dst, dinv, T, G, E);
  k_relu_bf16<<<gNF2, blk, 0, stream>>>(G, F1, NF / 2);

  // ---- damping MLP: 128->128 sig (bf16 out), 128->64 sig (f32 out), 64->1 sig
  k_gemm_bf16_wmma<128, 1, 1><<<gGemm, blk, 0, stream>>>(F1, 128, nullptr, 0, w_dp1, dp_b1, F0, N);
  k_gemm_bf16_wmma<64, 1, 0><<<gGemm, blk, 0, stream>>>(F0, 128, nullptr, 0, w_dp2, dp_b2, T, N);
  k_dp3<<<(N + 7) / 8, blk, 0, stream>>>(T, dp_W3, dp_b3, (float*)d_out, N);
}
